// MultichannelMultiheadAttention_56435870269523
// MI455X (gfx1250) — matrix operand order repaired and checked by reading, not compiled, not run on hardware
//
#include <hip/hip_runtime.h>
#include <hip/hip_bf16.h>

// ---------------------------------------------------------------------------
// MI455X (gfx1250) implementation.
// bf16 WMMA inputs, f32 accumulate (v_wmma_f32_16x16x32_bf16): 52 GFLOP total
// vs ~134MB minimum HBM traffic (~6us @ 23.3TB/s); bf16 K=32 WMMA keeps
// compute near the bandwidth floor. All operands pre-packed to bf16 once;
// GEMM tiles staged LDS-side by the Tensor Data Mover (tensor_load_to_lds,
// D# pad_enable for bank-conflict padding), double-buffered on TENSORcnt.
// Intermediates (<=33.5MB each) stay resident in the 192MB L2.
// ---------------------------------------------------------------------------

typedef __attribute__((ext_vector_type(16))) __bf16        v16bf;
typedef __attribute__((ext_vector_type(8)))  float         v8f;
typedef __attribute__((ext_vector_type(4)))  unsigned int  v4u;
typedef __attribute__((ext_vector_type(8)))  int           v8i;
typedef __attribute__((ext_vector_type(4)))  int           v4i;

#define BB    2
#define CC    8
#define BINS  1024
#define WW    512
#define HEADS 8
#define HDIM  128
#define ROTD  64

// ===========================================================================
// TDM: 2D tile (tile_rows x tile_cols bf16) from a row_stride-elem matrix
// into LDS at lds_addr, with 16B padding appended per 64B row (ISA 8.3/8.4:
// data_size=1(2B), pad_enable, pad_interval=3 (64B), pad_amount=3 (4 dwords)).
// This toolchain exposes the 6-arg builtin: (g0, g1, g2, g3, g4, cpol).
// ===========================================================================
__device__ __forceinline__ void tdm_load_tile_2d(unsigned lds_addr, const void* gptr,
                                                 unsigned tile_rows, unsigned tile_cols,
                                                 unsigned long long row_stride_elems) {
  const unsigned long long ga = (unsigned long long)(uintptr_t)gptr;
  v4u g0; v8i g1; v4i gz4; v8i gz8;
  // --- group 0: count=1 | lds_addr | global_addr[56:0] | type=2
  g0[0] = 1u;
  g0[1] = lds_addr;
  g0[2] = (unsigned)(ga & 0xffffffffu);
  g0[3] = (unsigned)((ga >> 32) & 0x1ffffffu) | (2u << 30);
  // --- group 1
  const unsigned td0 = 0x40000000u, td1 = 0x40000000u;   // huge dims: no OOB clip
  g1[0] = (int)((1u << 16) | (1u << 20) | (3u << 22) | (3u << 25));
  g1[1] = (int)((td0 & 0xffffu) << 16);                  // dim0[15:0]
  g1[2] = (int)((td0 >> 16) | ((td1 & 0xffffu) << 16));  // dim0[31:16] | dim1[15:0]
  g1[3] = (int)((td1 >> 16) | (tile_cols << 16));        // dim1[31:16] | tile_dim0
  g1[4] = (int)tile_rows;                                // tile_dim1 | tile_dim2=0
  g1[5] = (int)(unsigned)(row_stride_elems & 0xffffffffu);        // stride0 lo
  g1[6] = (int)(unsigned)((row_stride_elems >> 32) & 0xffffu);    // stride0 hi
  g1[7] = 0;
  gz4[0] = gz4[1] = gz4[2] = gz4[3] = 0;
  for (int i = 0; i < 8; ++i) gz8[i] = 0;
  __builtin_amdgcn_tensor_load_to_lds(g0, g1, gz4, gz4, gz8, 0);
}

// ===========================================================================
// Kernel: per-channel GEMM  Y[bc] = A[c] (1024x1024 bf16) @ B[bc] (1024x512)
// B supplied transposed: BT (bc, N=512, K=1024) bf16 row-major.
// Block tile 128x128, 8 waves 2(M)x4(N); K stepped by 32; A/B tiles staged by
// TDM into double-buffered padded LDS ([128][40] bf16).
// ===========================================================================
__global__ __launch_bounds__(256) void mc_gemm_bf16_kernel(
    const __bf16* __restrict__ Abf,   // (C, M, K)
    const __bf16* __restrict__ BTbf,  // (B*C, N, K)
    float*        __restrict__ Y)     // (B*C, M, N)
{
  const int M = BINS, N = WW, K = BINS;
  const int bc = blockIdx.z;
  const int ch = bc & (CC - 1);
  const __bf16* A  = Abf  + (size_t)ch * M * K;
  const __bf16* BT = BTbf + (size_t)bc * N * K;
  float*        Yg = Y    + (size_t)bc * M * N;
  const int m0 = blockIdx.y * 128;
  const int n0 = blockIdx.x * 128;

  __shared__ __bf16 Al[2][128][40];   // +8 bf16/row pad (TDM pad feature)
  __shared__ __bf16 Bl[2][128][40];

  const int tid = threadIdx.x;
  const int lane = tid & 31, wv = tid >> 5;
  const int l16 = lane & 15, lh = lane >> 4;
  const int wm = wv >> 2, wn = wv & 3;          // wave tile: 64(M) x 32(N)
  const bool loader = (wv == 0);                // wave 0 drives the TDM

  v8f acc[4][2];
  for (int mt = 0; mt < 4; ++mt)
    for (int nt = 0; nt < 2; ++nt)
      for (int r = 0; r < 8; ++r) acc[mt][nt][r] = 0.f;

  if (loader) {  // prime buffer 0
    tdm_load_tile_2d((unsigned)(uintptr_t)&Al[0][0][0], A + (size_t)m0 * K, 128, 32, K);
    tdm_load_tile_2d((unsigned)(uintptr_t)&Bl[0][0][0], BT + (size_t)n0 * K, 128, 32, K);
  }

  const int NIT = K / 32;   // 32
  for (int it = 0; it < NIT; ++it) {
    const int cur = it & 1;
    if (loader) __builtin_amdgcn_s_wait_tensorcnt(0);
    __syncthreads();                            // tile `cur` visible to all waves
    if (loader && (it + 1 < NIT)) {             // prefetch next tile into other buffer
      const int k1 = (it + 1) * 32;
      tdm_load_tile_2d((unsigned)(uintptr_t)&Al[1 - cur][0][0], A + (size_t)m0 * K + k1, 128, 32, K);
      tdm_load_tile_2d((unsigned)(uintptr_t)&Bl[1 - cur][0][0], BT + (size_t)n0 * K + k1, 128, 32, K);
    }

    // Fragments per ISA 7.12.2 layouts (b128-vectorizable LDS reads).
    v16bf af[4], bfr[2];
    for (int mt = 0; mt < 4; ++mt) {
      const __bf16* rp = &Al[cur][wm * 64 + mt * 16 + l16][0];
      for (int j = 0; j < 8; ++j) af[mt][j]     = rp[lh * 8 + j];       // K=lh*8+j
      for (int j = 0; j < 8; ++j) af[mt][8 + j] = rp[16 + lh * 8 + j];  // K=16+lh*8+j
    }
    for (int nt = 0; nt < 2; ++nt) {
      const __bf16* rp = &Bl[cur][wn * 32 + nt * 16 + l16][0];
      for (int j = 0; j < 8; ++j) bfr[nt][j]     = rp[lh * 8 + j];       // K=lh*8+j
      for (int j = 0; j < 8; ++j) bfr[nt][8 + j] = rp[16 + lh * 8 + j];  // K=16+lh*8+j
    }
    for (int mt = 0; mt < 4; ++mt)
      for (int nt = 0; nt < 2; ++nt)
        acc[mt][nt] = __builtin_amdgcn_wmma_f32_16x16x32_bf16(
            false, af[mt], false, bfr[nt], (short)0, acc[mt][nt], false, false);
  }

  // C layout: vgpr r -> rows r (lanes 0-15) / r+8 (lanes 16-31), n = lane%16
  for (int mt = 0; mt < 4; ++mt)
    for (int nt = 0; nt < 2; ++nt)
      for (int r = 0; r < 8; ++r) {
        const int mg = m0 + wm * 64 + mt * 16 + r + 8 * lh;
        const int ng = n0 + wn * 32 + nt * 16 + l16;
        Yg[(size_t)mg * N + ng] = acc[mt][nt][r];
      }
}

// ===========================================================================
// Pack kernels: one-time f32 -> bf16 (weights), and f32 -> bf16 transpose
// (activations, so GEMM B tiles are row-major for the TDM).
// ===========================================================================
__global__ __launch_bounds__(256) void pack_bf16_kernel(
    const float* __restrict__ in, __bf16* __restrict__ out, int n) {
  int i = blockIdx.x * 256 + threadIdx.x;
  if (i < n) out[i] = (__bf16)in[i];
}

__global__ __launch_bounds__(256) void pack_xT_kernel(
    const float* __restrict__ x,      // (B*C, 1024, 512)
    __bf16*      __restrict__ xT) {   // (B*C, 512, 1024)
  __shared__ __bf16 t[32][33];
  const int tid = threadIdx.x;
  const int bc = blockIdx.z, bin0 = blockIdx.y * 32, w0 = blockIdx.x * 32;
  const float* src = x + ((size_t)bc * BINS + bin0) * WW + w0;
  for (int i = 0; i < 4; ++i) {
    const int r = (tid >> 5) + 8 * i, c = tid & 31;
    t[r][c] = (__bf16)src[(size_t)r * WW + c];
  }
  __syncthreads();
  __bf16* dst = xT + ((size_t)bc * WW + w0) * BINS + bin0;
  for (int i = 0; i < 4; ++i) {
    const int r = (tid >> 5) + 8 * i, c = tid & 31;
    dst[(size_t)r * BINS + c] = t[c][r];
  }
}

// ===========================================================================
// 1x3 channel-mixing conv along W + bias; one thread per (b,bin,w) for all 8
// output channels (24-load reuse).
// ===========================================================================
__global__ __launch_bounds__(256) void conv_mix_kernel(
    const float* __restrict__ qp, const float* __restrict__ Wc,
    const float* __restrict__ bq, float* __restrict__ qc) {
  const int idx = blockIdx.x * 256 + threadIdx.x;
  const int w = idx & 511, bin = (idx >> 9) & 1023, b = idx >> 19;
  float in[CC][3];
  for (int i = 0; i < CC; ++i) {
    const float* p = qp + ((size_t)(b * CC + i) * BINS + bin) * WW + w;
    in[i][0] = (w > 0)      ? p[-1] : 0.f;
    in[i][1] = p[0];
    in[i][2] = (w < WW - 1) ? p[1]  : 0.f;
  }
  for (int o = 0; o < CC; ++o) {
    float a = bq[o];
    for (int i = 0; i < CC; ++i)
      for (int t = 0; t < 3; ++t)
        a = fmaf(in[i][t], Wc[(o * CC + i) * 3 + t], a);
    qc[((size_t)(b * CC + o) * BINS + bin) * WW + w] = a;
  }
}

// ===========================================================================
// Flash attention, one workgroup per (b,c,head). q == k == rotary(qh), v = qh,
// scale = 1/sqrt(1024). K (rotated) [s][d], V transposed [d][s], bf16 in LDS.
// Output written TRANSPOSED as bf16: aoT[bc][s][bin=h*128+d]  (coalesced, and
// directly the BT operand of the final GEMM). Dynamic LDS: 280,576 B.
// ===========================================================================
#define KSTR (HDIM + 8)    // 136
#define VSTR (WW + 8)      // 520

__global__ __launch_bounds__(256) void attn_flash_kernel(
    const float* __restrict__ qc,     // (B,C,1024,512)
    __bf16*      __restrict__ aoT) {  // (B*C, 512, 1024)
  extern __shared__ char smem[];
  __bf16* Klds = (__bf16*)smem;                 // [512][KSTR]
  __bf16* Vlds = Klds + WW * KSTR;              // [128][VSTR]
  __bf16* Pscr = Vlds + HDIM * VSTR;            // [8 waves][16][32]

  const int bch = blockIdx.x;                   // (b*C+c)*HEADS + h
  const int bc = bch >> 3, h = bch & 7;
  const float* qin = qc + (size_t)bch * HDIM * WW;     // (s,d) at [d*512+s]
  __bf16* aob = aoT + (size_t)bc * WW * BINS + h * HDIM;

  const int tid = threadIdx.x;

  // stage + rotary: pairs p (d=2p,2p+1), p<32 rotated; d>=64 passthrough
  for (int i = tid; i < 32 * WW; i += 256) {
    const int s = i & 511, p = i >> 9;
    const float x1 = qin[(2 * p) * WW + s];
    const float x2 = qin[(2 * p + 1) * WW + s];
    const float invf = __expf(-0.28782313f * (float)p);  // 10000^(-2p/64)
    float sn, cs;  __sincosf((float)s * invf, &sn, &cs);
    Klds[s * KSTR + 2 * p]     = (__bf16)(x1 * cs - x2 * sn);
    Klds[s * KSTR + 2 * p + 1] = (__bf16)(x2 * cs + x1 * sn);
    Vlds[(2 * p) * VSTR + s]     = (__bf16)x1;
    Vlds[(2 * p + 1) * VSTR + s] = (__bf16)x2;
  }
  for (int i = tid; i < 64 * WW; i += 256) {
    const int s = i & 511, d = ROTD + (i >> 9);
    const float v = qin[d * WW + s];
    Klds[s * KSTR + d] = (__bf16)v;
    Vlds[d * VSTR + s] = (__bf16)v;
  }
  __syncthreads();

  const int lane = tid & 31, wv = tid >> 5;
  const int l16 = lane & 15, lh = lane >> 4;
  __bf16* Pw = Pscr + wv * (16 * 32);
  const float scale = 0.03125f;

  for (int qt = 0; qt < 4; ++qt) {
    const int q0 = wv * 64 + qt * 16;
    v8f O[8];  float mrow[8], lrow[8];
    for (int t = 0; t < 8; ++t)
      for (int r = 0; r < 8; ++r) O[t][r] = 0.f;
    for (int r = 0; r < 8; ++r) { mrow[r] = -1e30f; lrow[r] = 0.f; }

    for (int ck = 0; ck < 16; ++ck) {
      const int s0 = ck * 32;
      v8f S0, S1;
      for (int r = 0; r < 8; ++r) { S0[r] = 0.f; S1[r] = 0.f; }

      for (int dc = 0; dc < 4; ++dc) {
        v16bf aq, b0, b1;
        const int ab = (q0 + l16) * KSTR + dc * 32;
        for (int j = 0; j < 8; ++j) aq[j]     = Klds[ab + lh * 8 + j];
        for (int j = 0; j < 8; ++j) aq[8 + j] = Klds[ab + 16 + lh * 8 + j];
        const int bb0 = (s0 + l16) * KSTR + dc * 32 + lh * 8;
        const int bb1 = (s0 + 16 + l16) * KSTR + dc * 32 + lh * 8;
        for (int e = 0; e < 16; ++e) { const int o = (e < 8) ? e : 8 + e; b0[e] = Klds[bb0 + o]; b1[e] = Klds[bb1 + o]; }
        S0 = __builtin_amdgcn_wmma_f32_16x16x32_bf16(false, aq, false, b0, (short)0, S0, false, false);
        S1 = __builtin_amdgcn_wmma_f32_16x16x32_bf16(false, aq, false, b1, (short)0, S1, false, false);
      }

      for (int r = 0; r < 8; ++r) {             // online softmax (rows r / r+8)
        const float v0 = S0[r] * scale, v1 = S1[r] * scale;
        float cm = fmaxf(v0, v1);
        for (int off = 8; off >= 1; off >>= 1) cm = fmaxf(cm, __shfl_xor(cm, off, 32));
        const float nm = fmaxf(mrow[r], cm);
        const float alpha = __expf(mrow[r] - nm);
        const float p0 = __expf(v0 - nm), p1 = __expf(v1 - nm);
        float ps = p0 + p1;
        for (int off = 8; off >= 1; off >>= 1) ps += __shfl_xor(ps, off, 32);
        lrow[r] = lrow[r] * alpha + ps;
        mrow[r] = nm;
        for (int t = 0; t < 8; ++t) O[t][r] *= alpha;
        const int m = r + 8 * lh;
        Pw[m * 32 + l16]      = (__bf16)p0;
        Pw[m * 32 + 16 + l16] = (__bf16)p1;
      }

      v16bf ap;                                 // P (16x32) @ V (32x128)
      const int pb = l16 * 32;
      for (int j = 0; j < 8; ++j) ap[j]     = Pw[pb + lh * 8 + j];
      for (int j = 0; j < 8; ++j) ap[8 + j] = Pw[pb + 16 + lh * 8 + j];
      for (int t2 = 0; t2 < 8; ++t2) {
        v16bf bv;
        const int vb = (t2 * 16 + l16) * VSTR + s0 + lh * 8;
        for (int e = 0; e < 16; ++e) bv[e] = Vlds[vb + ((e < 8) ? e : 8 + e)];
        O[t2] = __builtin_amdgcn_wmma_f32_16x16x32_bf16(false, ap, false, bv, (short)0, O[t2], false, false);
      }
    }

    // normalize & store transposed bf16 (lane-contiguous in d => coalesced)
    for (int r = 0; r < 8; ++r) {
      const float inv = 1.f / lrow[r];
      const int srow = q0 + r + 8 * lh;
      for (int t2 = 0; t2 < 8; ++t2) {
        const int d = t2 * 16 + l16;
        aob[(size_t)srow * BINS + d] = (__bf16)(O[t2][r] * inv);
      }
    }
  }
}

// ===========================================================================
extern "C" void kernel_launch(void* const* d_in, const int* in_sizes, int n_in,
                              void* d_out, int out_size, void* d_ws, size_t ws_size,
                              hipStream_t stream) {
  const float* x       = (const float*)d_in[0];
  const float* Wq_lin  = (const float*)d_in[1];
  const float* Wq_conv = (const float*)d_in[2];
  const float* bq_conv = (const float*)d_in[3];
  const float* Wout    = (const float*)d_in[4];
  float* out = (float*)d_out;

  const size_t TENS = (size_t)BB * CC * BINS * WW;     // 8,388,608
  const size_t WELEM = (size_t)CC * BINS * BINS;       // 8,388,608
  char* ws = (char*)d_ws;
  float*  qp  = (float*)ws;                            // 33.5 MB
  float*  qc  = (float*)(ws + TENS * 4);               // 33.5 MB
  __bf16* xT  = (__bf16*)(ws + TENS * 8);              // 16.8 MB (reused as aoT)
  __bf16* Wbf = (__bf16*)(ws + TENS * 8 + TENS * 2);   // 16.8 MB (reused for Wout)
  __bf16* aoT = xT;

  dim3 ggrid(WW / 128, BINS / 128, BB * CC);           // (4, 8, 16)
  const int pblk = (int)((WELEM + 255) / 256);

  // pack activations (transposed) + first weight to bf16
  pack_xT_kernel<<<dim3(WW / 32, BINS / 32, BB * CC), 256, 0, stream>>>(x, xT);
  pack_bf16_kernel<<<pblk, 256, 0, stream>>>(Wq_lin, Wbf, (int)WELEM);
  // 1) qp = Wq_lin[c] @ x[b,c]   (TDM-staged bf16 GEMM)
  mc_gemm_bf16_kernel<<<ggrid, 256, 0, stream>>>(Wbf, xT, qp);
  // pack second weight into the same buffer (Wq_lin copy now dead)
  pack_bf16_kernel<<<pblk, 256, 0, stream>>>(Wout, Wbf, (int)WELEM);
  // 2) qc = conv1x3(qp) + bias
  conv_mix_kernel<<<(int)(TENS / CC / 256), 256, 0, stream>>>(qp, Wq_conv, bq_conv, qc);
  // 3) flash attention -> aoT (bf16, transposed; overwrites xT which is dead)
  const size_t shmem = (size_t)(WW * KSTR + HDIM * VSTR + 8 * 16 * 32) * sizeof(__bf16);
  (void)hipFuncSetAttribute((const void*)attn_flash_kernel,
                            hipFuncAttributeMaxDynamicSharedMemorySize, (int)shmem);
  attn_flash_kernel<<<BB * CC * HEADS, 256, shmem, stream>>>(qc, aoT);
  // 4) out = Wout[c] @ attention
  mc_gemm_bf16_kernel<<<ggrid, 256, 0, stream>>>(Wbf, aoT, out);
}
